// BrokenBiasAttention_84430467105244
// MI455X (gfx1250) — matrix operand order repaired and checked by reading, not compiled, not run on hardware
//
#include <hip/hip_runtime.h>
#include <stdint.h>

// ---------- CDNA5 WMMA types ----------
typedef __bf16 bf16x16 __attribute__((ext_vector_type(16)));
typedef float  f32x8   __attribute__((ext_vector_type(8)));

union Frag { bf16x16 v; unsigned int u[8]; };

#define WMMA_BF16(A, B, Cacc) \
  __builtin_amdgcn_wmma_f32_16x16x32_bf16(false, (A), false, (B), (short)0, (Cacc), false, false)

__device__ __forceinline__ unsigned short f2bf(float f) {
  union { float f; unsigned int u; } v; v.f = f;
  unsigned int r = v.u + 0x7FFFu + ((v.u >> 16) & 1u);  // round-to-nearest-even
  return (unsigned short)(r >> 16);
}

// ---------- problem constants ----------
constexpr int BATCH = 4;
constexpr int NTOK  = 2048;          // 8*16*16
constexpr int CDIM  = 256;
constexpr int NH    = 8;
constexpr int HD    = 32;

// ============================================================
// fp32 -> bf16 flat convert (x)
// ============================================================
__global__ void cvt_bf16_kernel(const float* __restrict__ in,
                                unsigned short* __restrict__ out, int n) {
  int i = blockIdx.x * blockDim.x + threadIdx.x;
  if (i < n) out[i] = f2bf(in[i]);
}

// 256x256 fp32 -> bf16 transpose (weights become [N][K] so B-frag loads are dwords)
__global__ void transpose_w_kernel(const float* __restrict__ w,
                                   unsigned short* __restrict__ wt) {
  int i = blockIdx.x * blockDim.x + threadIdx.x;   // 65536 threads
  int k = i >> 8, c = i & 255;
  wt[c * 256 + k] = f2bf(w[k * 256 + c]);
}

// ============================================================
// QKV projection: one wave -> one 16x64 tile of X@W (K=256, 32 WMMAs),
// A-fragment reused across 4 column tiles.
// gridDim = (128 row-tiles * 4 col-groups, 3 matrices, 4 batches), block = 32
// Q stored [b,h,n,d] pre-scaled by 1/sqrt(32); K stored [b,h,n,d];
// V stored transposed [b,h,d,n] so P@V B-frags load as dwords.
// ============================================================
__global__ void __launch_bounds__(32)
qkv_gemm_kernel(const unsigned short* __restrict__ Xb,
                const unsigned short* __restrict__ Wqt,
                const unsigned short* __restrict__ Wkt,
                const unsigned short* __restrict__ Wvt,
                unsigned short* __restrict__ Qb,
                unsigned short* __restrict__ Kb,
                unsigned short* __restrict__ Vt) {
  const int lane = threadIdx.x & 31;
  const int half = lane >> 4, lr = lane & 15;
  const int which = blockIdx.y;        // 0=Q 1=K 2=V
  const int b     = blockIdx.z;
  const int ntile = blockIdx.x >> 2;   // 0..127
  const int cg    = blockIdx.x & 3;    // 64-wide column group

  const unsigned int* Au = (const unsigned int*)Xb + (size_t)(b * NTOK + ntile * 16) * 128;
  const unsigned short* Wt = (which == 0) ? Wqt : ((which == 1) ? Wkt : Wvt);
  const unsigned int* Bu = (const unsigned int*)Wt;   // [c][k], row stride 128 dwords

  const int cbase = cg * 64;
  Frag a, w;
  f32x8 acc[4] = {};
#pragma unroll
  for (int k0 = 0; k0 < 8; ++k0) {
#pragma unroll
    for (int j = 0; j < 8; ++j) {
      int kb = k0 * 32 + ((j & 3) * 2) + ((j >> 2) * 16) + half * 8;  // A-frag K base
      a.u[j] = Au[lr * 128 + (kb >> 1)];
    }
#pragma unroll
    for (int t = 0; t < 4; ++t) {
      int c = cbase + t * 16 + lr;
#pragma unroll
      for (int j = 0; j < 8; ++j) {
        int kk = k0 * 32 + ((j & 3) * 2) + ((j >> 2) * 16) + half * 8; // B-frag K base
        w.u[j] = Bu[c * 128 + (kk >> 1)];
      }
      acc[t] = WMMA_BF16(a.v, w.v, acc[t]);
    }
  }

  const float qscale = 0.17677669529663687f;   // 1/sqrt(HEAD)
#pragma unroll
  for (int t = 0; t < 4; ++t) {
    int c = cbase + t * 16 + lr;
    int h = c >> 5, d = c & 31;
#pragma unroll
    for (int j = 0; j < 8; ++j) {
      int n = ntile * 16 + j + half * 8;       // C-frag row
      if (which == 0)
        Qb[((size_t)((b * NH + h) * NTOK + n)) * HD + d] = f2bf(acc[t][j] * qscale);
      else if (which == 1)
        Kb[((size_t)((b * NH + h) * NTOK + n)) * HD + d] = f2bf(acc[t][j]);
      else
        Vt[((size_t)((b * NH + h) * HD + d)) * NTOK + n] = f2bf(acc[t][j]);
    }
  }
}

// ============================================================
// Flash attention: one wave per (b, head, 16-row tile).
// Per 32-column chunk: 2 WMMAs for S, on-the-fly bias gather,
// online softmax, P -> LDS -> A-frag, 2 WMMAs accumulating O.
// Next chunk's K/V rows are prefetched (global_prefetch_b8) to
// overlap memory latency with the softmax VALU work.
// ============================================================
__global__ void __launch_bounds__(32)
attn_kernel(const unsigned short* __restrict__ Qb,
            const unsigned short* __restrict__ Kb,
            const unsigned short* __restrict__ Vt,
            const float* __restrict__ bias_table,
            unsigned short* __restrict__ Ob) {
  __shared__ unsigned short Pl[16 * 32];
  const int lane = threadIdx.x & 31;
  const int half = lane >> 4, lr = lane & 15;
  const int ntile = blockIdx.x;   // 0..127
  const int h     = blockIdx.y;   // head
  const int b     = blockIdx.z;

  const unsigned int* Qu = (const unsigned int*)Qb + (size_t)((b * NH + h) * NTOK + ntile * 16) * 16;
  const unsigned int* Ku = (const unsigned int*)Kb + (size_t)(b * NH + h) * NTOK * 16;
  const unsigned int* Vu = (const unsigned int*)Vt + (size_t)(b * NH + h) * HD * 1024; // d stride 1024 dwords

  Frag aq, k0f, k1f, ap, v0f, v1f;
#pragma unroll
  for (int j = 0; j < 8; ++j) {
    int kb = ((j & 3) * 2) + ((j >> 2) * 16) + half * 8;
    aq.u[j] = Qu[lr * 16 + (kb >> 1)];
  }

  int rd1[8], rh1[8], rw1[8], nrow[8];
#pragma unroll
  for (int j = 0; j < 8; ++j) {
    int n = ntile * 16 + j + half * 8;
    nrow[j] = n;
    rd1[j] = n >> 8;            // depth coordinate (W=16,H=16)
    rh1[j] = (n >> 4) & 15;
    rw1[j] = n & 15;
  }

  float mi[8], li[8];
  f32x8 acc0 = {}, acc1 = {};
#pragma unroll
  for (int j = 0; j < 8; ++j) { mi[j] = -3.0e38f; li[j] = 0.0f; }
  const float* bt = bias_table + (size_t)h * 15 * 31 * 31;

#pragma unroll 1
  for (int mb = 0; mb < NTOK; mb += 32) {
    // K B-frags for this chunk
#pragma unroll
    for (int j = 0; j < 8; ++j) {
      int kk = ((j & 3) * 2) + ((j >> 2) * 16) + half * 8; // head-dim index pair
      k0f.u[j] = Ku[(mb + lr) * 16 + (kk >> 1)];
      k1f.u[j] = Ku[(mb + 16 + lr) * 16 + (kk >> 1)];
    }
    // prefetch next chunk's K rows and V rows (speculative, per-lane cachelines)
    if (mb + 32 < NTOK) {
      __builtin_prefetch((const void*)(Ku + (mb + 32 + lr) * 16), 0, 3);
      __builtin_prefetch((const void*)(Ku + (mb + 48 + lr) * 16), 0, 3);
      __builtin_prefetch((const void*)(Vu + lr * 1024 + ((mb + 32) >> 1)), 0, 3);
      __builtin_prefetch((const void*)(Vu + (16 + lr) * 1024 + ((mb + 32) >> 1)), 0, 3);
    }

    f32x8 z = {};
    f32x8 s0 = WMMA_BF16(aq.v, k0f.v, z);
    f32x8 s1 = WMMA_BF16(aq.v, k1f.v, z);

    // relative-position bias gather (table is L2-resident, 0.46 MB)
    int m0 = mb + lr, m1 = m0 + 16;
    int d2a = m0 >> 8, h2a = (m0 >> 4) & 15, w2a = m0 & 15;
    int d2b = m1 >> 8, h2b = (m1 >> 4) & 15, w2b = m1 & 15;
#pragma unroll
    for (int j = 0; j < 8; ++j) {
      int ia = ((rd1[j] - d2a + 7) * 31 + (rh1[j] - h2a + 15)) * 31 + (rw1[j] - w2a + 15);
      int ib = ((rd1[j] - d2b + 7) * 31 + (rh1[j] - h2b + 15)) * 31 + (rw1[j] - w2b + 15);
      s0[j] += bt[ia];
      s1[j] += bt[ib];
    }

    // online softmax (row = reg j + 8*half, replicated across 16 lanes)
#pragma unroll
    for (int j = 0; j < 8; ++j) {
      float t = fmaxf(s0[j], s1[j]);
      t = fmaxf(t, __shfl_xor(t, 1, 32));
      t = fmaxf(t, __shfl_xor(t, 2, 32));
      t = fmaxf(t, __shfl_xor(t, 4, 32));
      t = fmaxf(t, __shfl_xor(t, 8, 32));
      float mnew  = fmaxf(mi[j], t);
      float alpha = __expf(mi[j] - mnew);
      s0[j] = __expf(s0[j] - mnew);
      s1[j] = __expf(s1[j] - mnew);
      float rs = s0[j] + s1[j];
      rs += __shfl_xor(rs, 1, 32);
      rs += __shfl_xor(rs, 2, 32);
      rs += __shfl_xor(rs, 4, 32);
      rs += __shfl_xor(rs, 8, 32);
      li[j] = li[j] * alpha + rs;
      mi[j] = mnew;
      acc0[j] *= alpha;
      acc1[j] *= alpha;
    }

    // P (C-layout) -> LDS bf16 -> A-frag layout
#pragma unroll
    for (int j = 0; j < 8; ++j) {
      int row = j + half * 8;
      Pl[row * 32 + lr]      = f2bf(s0[j]);
      Pl[row * 32 + 16 + lr] = f2bf(s1[j]);
    }
    __syncthreads();
#pragma unroll
    for (int j = 0; j < 8; ++j) {
      int kb = ((j & 3) * 2) + ((j >> 2) * 16) + half * 8;
      ap.u[j] = ((const unsigned int*)Pl)[lr * 16 + (kb >> 1)];
      int kk = mb + ((j & 3) * 2) + ((j >> 2) * 16) + half * 8; // m index pair
      v0f.u[j] = Vu[lr * 1024 + (kk >> 1)];            // d = lr
      v1f.u[j] = Vu[(16 + lr) * 1024 + (kk >> 1)];     // d = 16+lr
    }
    acc0 = WMMA_BF16(ap.v, v0f.v, acc0);
    acc1 = WMMA_BF16(ap.v, v1f.v, acc1);
    __syncthreads();
  }

  // finalize: divide by row sums, store bf16 into [b, n, h*32+d]
#pragma unroll
  for (int j = 0; j < 8; ++j) {
    float inv = 1.0f / li[j];
    size_t base = ((size_t)(b * NTOK + nrow[j])) * CDIM + h * HD;
    Ob[base + lr]      = f2bf(acc0[j] * inv);
    Ob[base + 16 + lr] = f2bf(acc1[j] * inv);
  }
}

// ============================================================
// Output projection: one wave -> one 16x64 tile of O@Wo (fp32 out),
// A-fragment reused across 4 column tiles.
// ============================================================
__global__ void __launch_bounds__(32)
oproj_kernel(const unsigned short* __restrict__ Ob,
             const unsigned short* __restrict__ Wot,
             float* __restrict__ out) {
  const int lane = threadIdx.x & 31;
  const int half = lane >> 4, lr = lane & 15;
  const int rtile = blockIdx.x >> 2;           // 0..511
  const int cg    = blockIdx.x & 3;            // 64-wide column group

  const unsigned int* Au = (const unsigned int*)Ob + (size_t)rtile * 16 * 128;
  const unsigned int* Bu = (const unsigned int*)Wot;
  const int cbase = cg * 64;

  Frag a, w;
  f32x8 acc[4] = {};
#pragma unroll
  for (int k0 = 0; k0 < 8; ++k0) {
#pragma unroll
    for (int j = 0; j < 8; ++j) {
      int kb = k0 * 32 + ((j & 3) * 2) + ((j >> 2) * 16) + half * 8;
      a.u[j] = Au[lr * 128 + (kb >> 1)];
    }
#pragma unroll
    for (int t = 0; t < 4; ++t) {
      int c = cbase + t * 16 + lr;
#pragma unroll
      for (int j = 0; j < 8; ++j) {
        int kk = k0 * 32 + ((j & 3) * 2) + ((j >> 2) * 16) + half * 8;
        w.u[j] = Bu[c * 128 + (kk >> 1)];
      }
      acc[t] = WMMA_BF16(a.v, w.v, acc[t]);
    }
  }
#pragma unroll
  for (int t = 0; t < 4; ++t) {
    int c = cbase + t * 16 + lr;
#pragma unroll
    for (int j = 0; j < 8; ++j) {
      int row = rtile * 16 + j + half * 8;
      out[(size_t)row * 256 + c] = acc[t][j];
    }
  }
}

// ============================================================
extern "C" void kernel_launch(void* const* d_in, const int* in_sizes, int n_in,
                              void* d_out, int out_size, void* d_ws, size_t ws_size,
                              hipStream_t stream) {
  (void)in_sizes; (void)n_in; (void)out_size; (void)ws_size;
  const float* x    = (const float*)d_in[0];
  const float* Wq   = (const float*)d_in[1];
  const float* Wk   = (const float*)d_in[2];
  const float* Wv   = (const float*)d_in[3];
  const float* Wo   = (const float*)d_in[4];
  const float* bias = (const float*)d_in[5];
  float* out = (float*)d_out;

  // workspace layout (all bf16): ~17.3 MB total
  char* ws = (char*)d_ws;
  constexpr size_t SZ_X = (size_t)BATCH * NTOK * CDIM * 2;   // 4 MB
  constexpr size_t SZ_W = (size_t)CDIM * CDIM * 2;           // 128 KB
  unsigned short* Xb  = (unsigned short*)(ws);
  unsigned short* Wqt = (unsigned short*)(ws + SZ_X);
  unsigned short* Wkt = (unsigned short*)(ws + SZ_X + 1 * SZ_W);
  unsigned short* Wvt = (unsigned short*)(ws + SZ_X + 2 * SZ_W);
  unsigned short* Wot = (unsigned short*)(ws + SZ_X + 3 * SZ_W);
  unsigned short* Qb  = (unsigned short*)(ws + SZ_X + 4 * SZ_W);
  unsigned short* Kb  = (unsigned short*)(ws + SZ_X + 4 * SZ_W + 1 * SZ_X);
  unsigned short* Vt  = (unsigned short*)(ws + SZ_X + 4 * SZ_W + 2 * SZ_X);
  unsigned short* Ob  = Xb;   // reuse: X no longer needed after QKV projection

  // 1) convert / transpose inputs to bf16
  cvt_bf16_kernel<<<(BATCH * NTOK * CDIM) / 256, 256, 0, stream>>>(x, Xb, BATCH * NTOK * CDIM);
  transpose_w_kernel<<<256, 256, 0, stream>>>(Wq, Wqt);
  transpose_w_kernel<<<256, 256, 0, stream>>>(Wk, Wkt);
  transpose_w_kernel<<<256, 256, 0, stream>>>(Wv, Wvt);
  transpose_w_kernel<<<256, 256, 0, stream>>>(Wo, Wot);

  // 2) QKV projections (Q pre-scaled, V transposed), 16x64 tiles per wave
  qkv_gemm_kernel<<<dim3(512, 3, BATCH), 32, 0, stream>>>(Xb, Wqt, Wkt, Wvt, Qb, Kb, Vt);

  // 3) flash attention with fused relative-position bias + K/V prefetch
  attn_kernel<<<dim3(NTOK / 16, NH, BATCH), 32, 0, stream>>>(Qb, Kb, Vt, bias, Ob);

  // 4) output projection -> fp32, 16x64 tiles per wave
  oproj_kernel<<<(BATCH * NTOK / 16) * 4, 32, 0, stream>>>(Ob, Wot, out);
}
